// LSAM_26963804684381
// MI455X (gfx1250) — compile-verified, matrix operand order checked by reading, not run on hardware
//
#include <hip/hip_runtime.h>
#include <hip/hip_bf16.h>

// ---------------- CDNA5 / gfx1250 types ----------------
typedef __attribute__((ext_vector_type(16))) __bf16 v16bf;
typedef __attribute__((ext_vector_type(8)))  float  v8f;

#define S_LEN   2048
#define BATCH   16
#define DMODEL  512
#define NHEADS  8
#define LAYERS  2
#define DHEAD   64
#define KDIM    1024     // 2*D (xh width)
#define NQKV    1544     // 3*D + NH
#define NBIAS   1552     // padded to 16
#define NTILES  104      // 1664 padded cols / 16  (13 tiles per wave * 8 waves)
#define KSTEPS  32       // 1024 / 32

union BF16x16 {
    uint4          q[2];
    unsigned short u[16];
    v16bf          v;
};

__device__ __forceinline__ unsigned short f2bf(float f) {
    unsigned int x = __float_as_uint(f);
    x += 0x7FFFu + ((x >> 16) & 1u);           // round-to-nearest-even
    return (unsigned short)(x >> 16);
}

// K offset inside a 32-wide k-step for element `el` (0..15) of a lane's
// 16-bf16 WMMA operand fragment (mirrors the 16-bit A/B VGPR layout:
// lanes<16 hold K {0..7,16..23}, lanes>=16 hold K {8..15,24..31}).
__device__ __forceinline__ int klocal(int lane, int el) {
    int khalf = lane >> 4;
    int p = el >> 1, bit = el & 1;
    return ((p & 3) << 1) + bit + (khalf << 3) + ((p >= 4) ? 16 : 0);
}

// ---------------- workspace layout ----------------
static constexpr size_t alignUp(size_t v, size_t a) { return (v + a - 1) / a * a; }
constexpr size_t WB_ELEMS   = (size_t)LAYERS * NTILES * KSTEPS * 512;   // packed [Wqkv;Ww] bf16
constexpr size_t WOB_ELEMS  = (size_t)32 * 16 * 512;                    // packed Wo bf16
constexpr size_t BIAS_ELEMS = (size_t)LAYERS * NBIAS;
constexpr size_t FLAG_ELEMS = (size_t)S_LEN * BATCH;
constexpr size_t O_WB    = 0;
constexpr size_t O_WOB   = alignUp(O_WB   + WB_ELEMS  * 2, 256);
constexpr size_t O_BIAS  = alignUp(O_WOB  + WOB_ELEMS * 2, 256);
constexpr size_t O_OUT16 = alignUp(O_BIAS + BIAS_ELEMS * 4, 256);       // layer-1 h rows, bf16
constexpr size_t O_HMID  = alignUp(O_OUT16 + (size_t)S_LEN * BATCH * DMODEL * 2, 256);
constexpr size_t O_FLAGS = alignUp(O_HMID  + (size_t)S_LEN * BATCH * DMODEL * 2, 256);

// ---------------- kernel 1: pack weights into WMMA B-operand layout ----------------
__global__ void LSAM_prep_kernel(const float* __restrict__ Wqkv_w, const float* __restrict__ Wqkv_b,
                                 const float* __restrict__ Ww_w,   const float* __restrict__ Ww_b,
                                 const float* __restrict__ Wo_w,
                                 unsigned short* __restrict__ Wb, float* __restrict__ biasb,
                                 unsigned short* __restrict__ WoB, int* __restrict__ flags) {
    long long i = (long long)blockIdx.x * blockDim.x + threadIdx.x;
    if (i < (long long)WB_ELEMS) {
        int per_layer = NTILES * KSTEPS * 512;
        int layer = (int)(i / per_layer);
        int rem   = (int)(i % per_layer);
        int tile  = rem / (KSTEPS * 512);
        int r2    = rem % (KSTEPS * 512);
        int ks    = r2 / 512;
        int e     = r2 % 512;
        int lane  = e >> 4, el = e & 15;
        int n = tile * 16 + (lane & 15);
        int k = ks * 32 + klocal(lane, el);
        float val = 0.f;
        if (n < 1536)      val = Wqkv_w[((long long)layer * 1536 + n) * KDIM + k];
        else if (n < NQKV) val = Ww_w[((long long)layer * NHEADS + (n - 1536)) * KDIM + k];
        Wb[i] = f2bf(val);
        return;
    }
    i -= WB_ELEMS;
    if (i < (long long)WOB_ELEMS) {
        int tile = (int)(i / (16 * 512));
        int r2   = (int)(i % (16 * 512));
        int ks   = r2 / 512;
        int e    = r2 % 512;
        int lane = e >> 4, el = e & 15;
        int n = tile * 16 + (lane & 15);
        int k = ks * 32 + klocal(lane, el);
        WoB[i] = f2bf(Wo_w[(long long)n * DMODEL + k]);   // B[k][n] = Wo[n][k]
        return;
    }
    i -= WOB_ELEMS;
    if (i < (long long)BIAS_ELEMS) {
        int layer = (int)(i / NBIAS);
        int n     = (int)(i % NBIAS);
        float val = 0.f;
        if (n < 1536)      val = Wqkv_b[layer * 1536 + n];
        else if (n < NQKV) val = Ww_b[layer * NHEADS + (n - 1536)];
        biasb[i] = val;
        return;
    }
    i -= BIAS_ELEMS;
    if (i < (long long)FLAG_ELEMS) flags[i] = 0;
}

// ---------------- kernel 2: persistent recurrent scan ----------------
// grid = 32 blocks: block = (layer<<4) | batch. Fast-weight state AM lives in
// LDS for all 2048 steps (CDNA5: 320KB LDS/WGP). Layer pipeline via agent-scope
// flags: layer0 publishes h(t) -> layer1 consumes h(t) one stage behind.
__launch_bounds__(256, 1)
__global__ void LSAM_recurrent_kernel(const float* __restrict__ x,
                                      const unsigned short* __restrict__ Wb,
                                      const float* __restrict__ biasb,
                                      unsigned short* __restrict__ hmid,
                                      int* __restrict__ flags,
                                      unsigned short* __restrict__ outb16,
                                      float* __restrict__ dout) {
    __shared__ float AM[NHEADS][DHEAD][DHEAD + 1];     // 133 KB, +1 pad kills bank conflicts
    __shared__ float hstate[DMODEL];
    __shared__ __align__(16) unsigned short xh[KDIM];  // bf16 concat(input, h)
    __shared__ float qkvw[NBIAS];
    __shared__ float knorm[NHEADS][DHEAD];
    __shared__ float qnorm[NHEADS][DHEAD];

    const int tid  = threadIdx.x;
    const int lane = tid & 31;
    const int wave = tid >> 5;
    const int b     = blockIdx.x & 15;
    const int layer = blockIdx.x >> 4;

    for (int i = tid; i < DMODEL; i += 256) hstate[i] = 0.f;
    float* amf = &AM[0][0][0];
    for (int i = tid; i < NHEADS * DHEAD * (DHEAD + 1); i += 256) amf[i] = 0.f;
    __syncthreads();

    const unsigned short* wbL   = Wb + (long long)layer * NTILES * KSTEPS * 512;
    const float*          biasL = biasb + layer * NBIAS;

    for (int t = 0; t < S_LEN; ++t) {
        // ---- build xh = bf16(concat(input, own h)) ----
        if (layer == 0) {
            const float* xin = x + ((long long)t * BATCH + b) * DMODEL;
            xh[tid]       = f2bf(xin[tid]);
            xh[tid + 256] = f2bf(xin[tid + 256]);
        } else {
            if (tid == 0) {
                while (__hip_atomic_load(&flags[t * BATCH + b], __ATOMIC_ACQUIRE,
                                         __HIP_MEMORY_SCOPE_AGENT) == 0)
                    __builtin_amdgcn_s_sleep(1);
            }
            __syncthreads();
            const unsigned short* hin = hmid + ((long long)t * BATCH + b) * DMODEL;
            xh[tid]       = hin[tid];
            xh[tid + 256] = hin[tid + 256];
        }
        xh[512 + tid] = f2bf(hstate[tid]);
        xh[768 + tid] = f2bf(hstate[tid + 256]);
        __syncthreads();

        // ---- qkv/w projection: (1x1024)x(1024x1664) GEMV on the WMMA path ----
        {
            v8f acc[13];
            const v8f vzero = {0.f, 0.f, 0.f, 0.f, 0.f, 0.f, 0.f, 0.f};
#pragma unroll
            for (int i = 0; i < 13; ++i) acc[i] = vzero;
            const unsigned short* wbW = wbL + (long long)(wave * 13) * KSTEPS * 512;
            for (int ks = 0; ks < KSTEPS; ++ks) {
                BF16x16 af;                              // A: xh in M-row 0 only
                if ((lane & 15) == 0) {
                    int base = ks * 32 + (lane >> 4) * 8;
                    af.q[0] = *(const uint4*)&xh[base];
                    af.q[1] = *(const uint4*)&xh[base + 16];
                } else {
                    af.q[0] = make_uint4(0, 0, 0, 0);
                    af.q[1] = make_uint4(0, 0, 0, 0);
                }
                const unsigned short* wp = wbW + ks * 512 + lane * 16;
                if (ks + 1 < KSTEPS) __builtin_prefetch(wp + 512, 0, 0);
#pragma unroll
                for (int i = 0; i < 13; ++i) {
                    const uint4* p = (const uint4*)(wp + (long long)i * KSTEPS * 512);
                    BF16x16 bf_;
                    bf_.q[0] = p[0];
                    bf_.q[1] = p[1];
                    acc[i] = __builtin_amdgcn_wmma_f32_16x16x32_bf16(
                        false, af.v, false, bf_.v, (short)0, acc[i], false, false);
                }
            }
            if (lane < 16) {                             // D row M=0 lives in VGPR0, lanes 0-15
#pragma unroll
                for (int i = 0; i < 13; ++i) {
                    int n = (wave * 13 + i) * 16 + lane;
                    if (n < NBIAS) qkvw[n] = acc[i][0] + biasL[n];
                }
            }
        }
        __syncthreads();

        // ---- per-head fast-weight update (wave == head) ----
        {
            const int h = wave;
            const int v0 = lane, v1 = lane + 32;
            float q0 = qkvw[h * 64 + v0],        q1 = qkvw[h * 64 + v1];
            float k0 = qkvw[512 + h * 64 + v0],  k1 = qkvw[512 + h * 64 + v1];
            float sq = q0 * q0 + q1 * q1, sk = k0 * k0 + k1 * k1;
#pragma unroll
            for (int m = 16; m >= 1; m >>= 1) {
                sq += __shfl_xor(sq, m, 32);
                sk += __shfl_xor(sk, m, 32);
            }
            float rq = 1.f / fmaxf(sqrtf(sq), 1e-12f);
            float rk = 1.f / fmaxf(sqrtf(sk), 1e-12f);
            float qn0 = q0 * rq, qn1 = q1 * rq, kn0 = k0 * rk, kn1 = k1 * rk;
            qnorm[h][v0] = qn0; qnorm[h][v1] = qn1;
            knorm[h][v0] = kn0; knorm[h][v1] = kn1;
            float kq = kn0 * qn0 + kn1 * qn1;            // dot(k_n, q_n)
#pragma unroll
            for (int m = 16; m >= 1; m >>= 1) kq += __shfl_xor(kq, m, 32);
            __syncthreads();

            // pass 1 over AM (old): v_r = AM@k and AM@q fused
            float acck0 = 0.f, accq0 = 0.f, acck1 = 0.f, accq1 = 0.f;
#pragma unroll 8
            for (int qi = 0; qi < DHEAD; ++qi) {
                float kk = knorm[h][qi], qq = qnorm[h][qi];
                float a0 = AM[h][v0][qi], a1 = AM[h][v1][qi];
                acck0 = fmaf(a0, kk, acck0); accq0 = fmaf(a0, qq, accq0);
                acck1 = fmaf(a1, kk, acck1); accq1 = fmaf(a1, qq, accq1);
            }
            float wsig = 1.f / (1.f + __expf(-qkvw[1536 + h]));
            float vv0 = qkvw[1024 + h * 64 + v0], vv1 = qkvw[1024 + h * 64 + v1];
            float d0 = wsig * (vv0 - acck0), d1 = wsig * (vv1 - acck1);
            // pass 2: rank-1 update AM += delta (x) k_n
#pragma unroll 8
            for (int qi = 0; qi < DHEAD; ++qi) {
                float kk = knorm[h][qi];
                AM[h][v0][qi] += d0 * kk;
                AM[h][v1][qi] += d1 * kk;
            }
            // h_new = AM_new@q = AM_old@q + delta*(k.q)
            hstate[h * 64 + v0] = accq0 + d0 * kq;
            hstate[h * 64 + v1] = accq1 + d1 * kq;
        }
        __syncthreads();

        // ---- publish ----
        if (layer == 0) {
            unsigned short* row = hmid + ((long long)t * BATCH + b) * DMODEL;
            row[tid]       = f2bf(hstate[tid]);
            row[tid + 256] = f2bf(hstate[tid + 256]);
            __threadfence();
            __syncthreads();
            if (tid == 0)
                __hip_atomic_store(&flags[t * BATCH + b], 1, __ATOMIC_RELEASE,
                                   __HIP_MEMORY_SCOPE_AGENT);
        } else {
            unsigned short* row = outb16 + ((long long)t * BATCH + b) * DMODEL;
            row[tid]       = f2bf(hstate[tid]);
            row[tid + 256] = f2bf(hstate[tid + 256]);
        }
        __syncthreads();
    }

    // ---- final states (last layer) into d_out tail ----
    if (layer == LAYERS - 1) {
        const long long YOFF = (long long)S_LEN * BATCH * DMODEL;
        for (int i = tid; i < DMODEL; i += 256)
            dout[YOFF + (long long)b * DMODEL + i] = hstate[i];
        const long long AOFF = YOFF + (long long)BATCH * DMODEL;
        for (int i = tid; i < NHEADS * DHEAD * DHEAD; i += 256) {
            int hh = i >> 12, rem = i & 4095, v = rem >> 6, qq = rem & 63;
            dout[AOFF + (long long)b * NHEADS * DHEAD * DHEAD + i] = AM[hh][v][qq];
        }
    }
}

// ---------------- kernel 3: out @ Wo^T + bias -> relu -> +x -> LayerNorm ----------------
// Dense WMMA GEMM: M = S*B = 32768 rows. Block = 16 rows x full N=512 so the
// LayerNorm row reduction stays in-block. 8 waves: wave w owns N[64w, 64w+64).
__launch_bounds__(256, 2)
__global__ void LSAM_out_kernel(const unsigned short* __restrict__ outb16,
                                const unsigned short* __restrict__ WoB,
                                const float* __restrict__ Wo_b,
                                const float* __restrict__ x,
                                const float* __restrict__ ln_g,
                                const float* __restrict__ ln_b,
                                float* __restrict__ y) {
    __shared__ float ytile[16][DMODEL + 4];
    const int tid = threadIdx.x, lane = tid & 31, wave = tid >> 5;
    const long long r0 = (long long)blockIdx.x * 16;

    v8f acc[4];
    const v8f vzero = {0.f, 0.f, 0.f, 0.f, 0.f, 0.f, 0.f, 0.f};
#pragma unroll
    for (int nt = 0; nt < 4; ++nt) acc[nt] = vzero;

    const long long arow = r0 + (lane & 15);
    for (int ks = 0; ks < 16; ++ks) {
        int base = ks * 32 + (lane >> 4) * 8;
        BF16x16 af;
        af.q[0] = *(const uint4*)(outb16 + arow * DMODEL + base);
        af.q[1] = *(const uint4*)(outb16 + arow * DMODEL + base + 16);
#pragma unroll
        for (int nt = 0; nt < 4; ++nt) {
            int tile = wave * 4 + nt;
            const uint4* p = (const uint4*)(WoB + ((long long)tile * 16 + ks) * 512 + lane * 16);
            BF16x16 bf_;
            bf_.q[0] = p[0];
            bf_.q[1] = p[1];
            acc[nt] = __builtin_amdgcn_wmma_f32_16x16x32_bf16(
                false, af.v, false, bf_.v, (short)0, acc[nt], false, false);
        }
    }

    // epilogue: bias + relu + residual into LDS (C/D layout: VGPR j -> M=j+8*(lane/16))
#pragma unroll
    for (int nt = 0; nt < 4; ++nt) {
        int n = wave * 64 + nt * 16 + (lane & 15);
        float bias = Wo_b[n];
#pragma unroll
        for (int j = 0; j < 8; ++j) {
            int m = j + 8 * (lane >> 4);
            long long r = r0 + m;
            ytile[m][n] = fmaxf(acc[nt][j] + bias, 0.f) + x[r * DMODEL + n];
        }
    }
    __syncthreads();

    // LayerNorm: wave handles rows 2w, 2w+1
#pragma unroll
    for (int rr = 0; rr < 2; ++rr) {
        int m = wave * 2 + rr;
        long long r = r0 + m;
        float s = 0.f;
#pragma unroll
        for (int i = 0; i < 16; ++i) s += ytile[m][lane + 32 * i];
#pragma unroll
        for (int mm = 16; mm >= 1; mm >>= 1) s += __shfl_xor(s, mm, 32);
        float mu = s * (1.f / 512.f);
        float vs = 0.f;
#pragma unroll
        for (int i = 0; i < 16; ++i) {
            float d = ytile[m][lane + 32 * i] - mu;
            vs = fmaf(d, d, vs);
        }
#pragma unroll
        for (int mm = 16; mm >= 1; mm >>= 1) vs += __shfl_xor(vs, mm, 32);
        float rstd = rsqrtf(vs * (1.f / 512.f) + 1e-5f);
#pragma unroll
        for (int i = 0; i < 16; ++i) {
            int n = lane + 32 * i;
            y[r * DMODEL + n] = (ytile[m][n] - mu) * rstd * ln_g[n] + ln_b[n];
        }
    }
}

// ---------------- host launcher ----------------
extern "C" void kernel_launch(void* const* d_in, const int* in_sizes, int n_in,
                              void* d_out, int out_size, void* d_ws, size_t ws_size,
                              hipStream_t stream) {
    const float* x      = (const float*)d_in[0];
    const float* Wqkv_w = (const float*)d_in[1];
    const float* Wqkv_b = (const float*)d_in[2];
    const float* Ww_w   = (const float*)d_in[3];
    const float* Ww_b   = (const float*)d_in[4];
    const float* Wo_w   = (const float*)d_in[5];
    const float* Wo_b   = (const float*)d_in[6];
    const float* ln_g   = (const float*)d_in[7];
    const float* ln_b   = (const float*)d_in[8];
    (void)in_sizes; (void)n_in; (void)out_size; (void)ws_size;

    char* ws = (char*)d_ws;
    unsigned short* Wb     = (unsigned short*)(ws + O_WB);
    unsigned short* WoB    = (unsigned short*)(ws + O_WOB);
    float*          biasb  = (float*)(ws + O_BIAS);
    unsigned short* outb16 = (unsigned short*)(ws + O_OUT16);
    unsigned short* hmid   = (unsigned short*)(ws + O_HMID);
    int*            flags  = (int*)(ws + O_FLAGS);

    const long long prep_total = (long long)(WB_ELEMS + WOB_ELEMS + BIAS_ELEMS + FLAG_ELEMS);
    int prep_blocks = (int)((prep_total + 255) / 256);
    LSAM_prep_kernel<<<prep_blocks, 256, 0, stream>>>(Wqkv_w, Wqkv_b, Ww_w, Ww_b, Wo_w,
                                                      Wb, biasb, WoB, flags);

    LSAM_recurrent_kernel<<<LAYERS * BATCH, 256, 0, stream>>>(
        x, Wb, biasb, hmid, flags, outb16, (float*)d_out);

    LSAM_out_kernel<<<(S_LEN * BATCH) / 16, 256, 0, stream>>>(
        outb16, WoB, Wo_b, x, ln_g, ln_b, (float*)d_out);
}
